// SpectralDependencyModule_78769700208952
// MI455X (gfx1250) — compile-verified, matrix operand order checked by reading, not run on hardware
//
#include <hip/hip_runtime.h>
#include <math.h>

// Problem constants (from reference): B=4, C=128, H=W=64 -> N=4096
#define B_ 4
#define C_ 128
#define N_ 4096

typedef _Float16 h8   __attribute__((ext_vector_type(8)));
typedef _Float16 v16h __attribute__((ext_vector_type(16)));
typedef float    v8f  __attribute__((ext_vector_type(8)));
typedef unsigned int u32x4 __attribute__((ext_vector_type(4)));
typedef int          i32x4 __attribute__((ext_vector_type(4)));
typedef int          i32x8 __attribute__((ext_vector_type(8)));

// Load a WMMA 16-bit A/B fragment (ISA 7.12.2 layout): per lane, two contiguous
// 8 x f16 chunks at +sel and +16+sel (sel = 0 for lanes 0-15, 8 for lanes 16-31).
__device__ __forceinline__ v16h load_frag(const _Float16* p, int sel) {
  h8 lo = *(const h8*)(p + sel);
  h8 hi = *(const h8*)(p + 16 + sel);
  return __builtin_shufflevector(lo, hi, 0,1,2,3,4,5,6,7,8,9,10,11,12,13,14,15);
}

__device__ __forceinline__ v8f wmma_f16(v16h a, v16h b, v8f c) {
  return __builtin_amdgcn_wmma_f32_16x16x32_f16(false, a, false, b, (short)0, c,
                                                false, false);
}

// ---------------------------------------------------------------------------
// TDM: async-load a 32-row x 256-byte 2D tile (32 x 128 f16, row-major,
// contiguous) from global memory into LDS. D# built per ISA ch.8:
//   group0: count=1 | lds_addr | global_addr[56:0] | type=2
//   group1: data_size=3 (8B units), tensor/tile dims = 32x32 units,
//           tensor_dim0_stride = 32 units (256B row pitch)
// Tracked on TENSORcnt. Issue once per tile from a single wave.
// ---------------------------------------------------------------------------
__device__ __forceinline__ void tdm_load_tile_32x256B(unsigned lds_off,
                                                      const _Float16* gsrc) {
  unsigned long long ga = (unsigned long long)(size_t)gsrc;
  u32x4 g0;
  g0[0] = 1u;                                   // count=1 (valid user D#)
  g0[1] = lds_off;                              // lds_addr (bytes)
  g0[2] = (unsigned)ga;                         // global_addr[31:0]
  g0[3] = (unsigned)((ga >> 32) & 0x01FFFFFFu)  // global_addr[56:32]
          | (2u << 30);                         // type=2 ("image")
  i32x8 g1;
  g1[0] = (3 << 16);        // workgroup_mask=0, data_size=3 (8B)
  g1[1] = (32 << 16);       // tensor_dim0 = 32 units (bits 79:48)
  g1[2] = (32 << 16);       // tensor_dim1 = 32 rows (bits 111:80)
  g1[3] = (32 << 16);       // tile_dim0  = 32 units (bits 127:112)
  g1[4] = 32;               // tile_dim1  = 32 rows  (bits 143:128)
  g1[5] = 32;               // tensor_dim0_stride = 32 units (bits 207:160)
  g1[6] = 0;
  g1[7] = 0;
  i32x4 z4 = {0, 0, 0, 0};  // 2-D tensor: groups 2/3 unused
#if __clang_major__ >= 23
  i32x8 z8 = {0, 0, 0, 0, 0, 0, 0, 0};
  __builtin_amdgcn_tensor_load_to_lds(g0, g1, z4, z4, z8, 0);
#else
  __builtin_amdgcn_tensor_load_to_lds(g0, g1, z4, z4, 0);
#endif
}

// ---------------------------------------------------------------------------
// Kernel A: x[B,C,N] fp32 -> t[B,N,C] f16 (tiled transpose through LDS)
// ---------------------------------------------------------------------------
__global__ void __launch_bounds__(256)
transpose_to_f16(const float* __restrict__ x, _Float16* __restrict__ t) {
  __shared__ float tile[32][33];
  int b  = blockIdx.z;
  int c0 = blockIdx.y * 32;
  int n0 = blockIdx.x * 32;
  int tx = threadIdx.x, ty = threadIdx.y;
  const float* xp = x + (size_t)b * C_ * N_;
  #pragma unroll
  for (int i = 0; i < 4; ++i)
    tile[ty + 8*i][tx] = xp[(size_t)(c0 + ty + 8*i) * N_ + n0 + tx];
  __syncthreads();
  _Float16* tp = t + (size_t)b * N_ * C_;
  #pragma unroll
  for (int i = 0; i < 4; ++i)
    tp[(size_t)(n0 + ty + 8*i) * C_ + c0 + tx] = (_Float16)tile[tx][ty + 8*i];
}

// ---------------------------------------------------------------------------
// Kernel B: QKV projection. Q[b,n,d] = sum_c t[b,n,c] * W[d,c]
// Direct kernarg output pointer per call -> global_store_b16 (not flat).
// ---------------------------------------------------------------------------
__device__ __forceinline__ void proj_one(const float* __restrict__ W,
                                         _Float16* __restrict__ out,
                                         _Float16* Wlds, const v16h aq[4],
                                         int n0, int wave, int lane16,
                                         int sel, int hi8) {
  __syncthreads();
  const float4* src = (const float4*)W;
  for (int i = threadIdx.x; i < C_ * C_ / 4; i += 256) {
    float4 v = src[i];
    Wlds[4*i+0] = (_Float16)v.x; Wlds[4*i+1] = (_Float16)v.y;
    Wlds[4*i+2] = (_Float16)v.z; Wlds[4*i+3] = (_Float16)v.w;
  }
  __syncthreads();
  #pragma unroll
  for (int tcol = 0; tcol < 8; ++tcol) {
    int d0 = tcol * 16;
    const _Float16* wrow = &Wlds[(d0 + lane16) * C_];
    v8f acc = {};
    #pragma unroll
    for (int kk = 0; kk < 4; ++kk) {
      v16h bw = load_frag(wrow + 32 * kk, sel);
      acc = wmma_f16(aq[kk], bw, acc);
    }
    #pragma unroll
    for (int r = 0; r < 8; ++r)
      out[(size_t)(n0 + wave*16 + r + hi8) * C_ + d0 + lane16] = (_Float16)acc[r];
  }
}

__global__ void __launch_bounds__(256)
qkv_proj(const _Float16* __restrict__ t,
         const float* __restrict__ Wq, const float* __restrict__ Wk,
         const float* __restrict__ Wv,
         _Float16* __restrict__ Qh, _Float16* __restrict__ Kh,
         _Float16* __restrict__ Vh) {
  __shared__ __align__(16) _Float16 Wlds[C_ * C_];   // 32 KB
  int b    = blockIdx.y;
  int n0   = blockIdx.x * 128;
  int lane = threadIdx.x & 31;
  int wave = threadIdx.x >> 5;
  int lane16 = lane & 15;
  int sel  = (lane < 16) ? 0 : 8;
  int hi8  = (lane < 16) ? 0 : 8;

  const _Float16* trow = t + ((size_t)b * N_ + n0 + wave * 16 + lane16) * C_;
  v16h aq[4];
  #pragma unroll
  for (int kk = 0; kk < 4; ++kk) aq[kk] = load_frag(trow + 32 * kk, sel);

  size_t boff = (size_t)b * N_ * C_;
  proj_one(Wq, Qh + boff, Wlds, aq, n0, wave, lane16, sel, hi8);
  proj_one(Wk, Kh + boff, Wlds, aq, n0, wave, lane16, sel, hi8);
  proj_one(Wv, Vh + boff, Wlds, aq, n0, wave, lane16, sel, hi8);
}

// ---------------------------------------------------------------------------
// Kernel C: flash attention, TDM double-buffered K/V staging.
// Per block: 128 query rows of one batch; loop over 32-key tiles.
// ---------------------------------------------------------------------------
__global__ void __launch_bounds__(256)
flash_attn(const _Float16* __restrict__ Qh, const _Float16* __restrict__ Kh,
           const _Float16* __restrict__ Vh, float* __restrict__ out) {
  __shared__ __align__(16) _Float16 Klds[2][32 * C_];   // [buf][key][c]  16 KB
  __shared__ __align__(16) _Float16 Vlds[2][32 * C_];   // [buf][key][c]  16 KB
  __shared__ __align__(16) _Float16 VTlds[2][C_ * 32];  // [buf][c][key]  16 KB
  __shared__ __align__(16) _Float16 Plds[8][16 * 32];   // per-wave P      8 KB

  int b    = blockIdx.y;
  int q0   = blockIdx.x * 128;
  int lane = threadIdx.x & 31;
  int wave = threadIdx.x >> 5;
  int lane16 = lane & 15;
  int sel  = (lane < 16) ? 0 : 8;
  int hi8  = (lane < 16) ? 0 : 8;

  const float scale = 0.015625f;           // quirk: 1/sqrt(N) = 1/64
  const float LOG2E = 1.44269504088896f;

  // Q fragments held in registers for the whole kernel
  const _Float16* qrow = Qh + ((size_t)b * N_ + q0 + wave*16 + lane16) * C_;
  v16h aq[4];
  #pragma unroll
  for (int kk = 0; kk < 4; ++kk) aq[kk] = load_frag(qrow + 32 * kk, sel);

  v8f o[8];
  #pragma unroll
  for (int tc = 0; tc < 8; ++tc) o[tc] = (v8f){};
  float m_r[8], l_r[8];
  #pragma unroll
  for (int r = 0; r < 8; ++r) { m_r[r] = -__builtin_inff(); l_r[r] = 0.f; }

  const _Float16* Kb = Qh; // placeholder to keep types; set properly below
  Kb = Kh + (size_t)b * N_ * C_;
  const _Float16* Vb = Vh + (size_t)b * N_ * C_;

  unsigned koff[2], voff[2];
  koff[0] = (unsigned)(size_t)(void*)&Klds[0][0];
  koff[1] = (unsigned)(size_t)(void*)&Klds[1][0];
  voff[0] = (unsigned)(size_t)(void*)&Vlds[0][0];
  voff[1] = (unsigned)(size_t)(void*)&Vlds[1][0];

  // Prologue: async-stage tile 0 (one TDM issue per tile, wave 0 only)
  if (wave == 0) {
    tdm_load_tile_32x256B(koff[0], Kb);
    tdm_load_tile_32x256B(voff[0], Vb);
  }

  int cur = 0;
  for (int j = 0; j < N_; j += 32) {
    if (wave == 0) __builtin_amdgcn_s_wait_tensorcnt(0);
    __syncthreads();   // tile `cur` ready for everyone; prev compute finished

    // Kick off the next tile into the other buffer while we compute
    if (j + 32 < N_ && wave == 0) {
      tdm_load_tile_32x256B(koff[cur ^ 1], Kb + (size_t)(j + 32) * C_);
      tdm_load_tile_32x256B(voff[cur ^ 1], Vb + (size_t)(j + 32) * C_);
    }

    // Build V^T (LDS->LDS) so P*V B-fragments are contiguous 128-bit reads
    {
      int kr = threadIdx.x >> 3;
      int cg = (threadIdx.x & 7) * 16;
      const _Float16* src = &Vlds[cur][kr * C_ + cg];
      h8 v0 = *(const h8*)(src);
      h8 v1 = *(const h8*)(src + 8);
      #pragma unroll
      for (int i = 0; i < 8; ++i) VTlds[cur][(cg + i) * 32 + kr] = v0[i];
      #pragma unroll
      for (int i = 0; i < 8; ++i) VTlds[cur][(cg + 8 + i) * 32 + kr] = v1[i];
    }
    __syncthreads();   // V^T visible block-wide

    // S = Q * K^T for two 16-key subtiles
    const _Float16* Kt = &Klds[cur][0];
    v8f s0 = (v8f){}, s1 = (v8f){};
    #pragma unroll
    for (int kk = 0; kk < 4; ++kk) {
      v16h bk0 = load_frag(Kt + lane16 * C_ + 32 * kk, sel);
      v16h bk1 = load_frag(Kt + (16 + lane16) * C_ + 32 * kk, sel);
      s0 = wmma_f16(aq[kk], bk0, s0);
      s1 = wmma_f16(aq[kk], bk1, s1);
    }

    // Online softmax. Accumulator row r lives across lanes 0-15 (rows 0-7)
    // and lanes 16-31 (rows 8-15) -> width-16 shuffle reductions.
    float p0[8], p1[8];
    #pragma unroll
    for (int r = 0; r < 8; ++r) {
      float a  = s0[r] * scale;
      float c2 = s1[r] * scale;
      float mx = fmaxf(a, c2);
      #pragma unroll
      for (int off = 8; off >= 1; off >>= 1)
        mx = fmaxf(mx, __shfl_xor(mx, off, 16));
      float m_new = fmaxf(m_r[r], mx);
      float alpha = exp2f((m_r[r] - m_new) * LOG2E);
      float e0 = exp2f((a  - m_new) * LOG2E);
      float e1 = exp2f((c2 - m_new) * LOG2E);
      float rs = e0 + e1;
      #pragma unroll
      for (int off = 8; off >= 1; off >>= 1)
        rs += __shfl_xor(rs, off, 16);
      l_r[r] = l_r[r] * alpha + rs;
      m_r[r] = m_new;
      p0[r] = e0; p1[r] = e1;
      #pragma unroll
      for (int tc = 0; tc < 8; ++tc) o[tc][r] = o[tc][r] * alpha;
    }

    // D-layout P -> A-fragment layout via per-wave LDS scratch
    _Float16* P = &Plds[wave][0];
    #pragma unroll
    for (int r = 0; r < 8; ++r) {
      P[(r + hi8) * 32 + lane16]      = (_Float16)p0[r];
      P[(r + hi8) * 32 + 16 + lane16] = (_Float16)p1[r];
    }
    asm volatile("s_wait_dscnt 0" ::: "memory");   // CDNA5 split-counter wait
    v16h ap = load_frag(P + lane16 * 32, sel);

    // O += P * V
    const _Float16* Vt = &VTlds[cur][0];
    #pragma unroll
    for (int tc = 0; tc < 8; ++tc) {
      v16h bv = load_frag(Vt + (tc * 16 + lane16) * 32, sel);
      o[tc] = wmma_f16(ap, bv, o[tc]);
    }
    cur ^= 1;
  }

  // Normalize and store fp32 output, flat [B][N][C] (matches reference view quirk)
  float* ob = out + (size_t)b * N_ * C_;
  #pragma unroll
  for (int tc = 0; tc < 8; ++tc) {
    #pragma unroll
    for (int r = 0; r < 8; ++r) {
      float inv = 1.0f / l_r[r];
      ob[(size_t)(q0 + wave*16 + r + hi8) * C_ + tc*16 + lane16] = o[tc][r] * inv;
    }
  }
}

// ---------------------------------------------------------------------------
extern "C" void kernel_launch(void* const* d_in, const int* in_sizes, int n_in,
                              void* d_out, int out_size, void* d_ws, size_t ws_size,
                              hipStream_t stream) {
  const float* x  = (const float*)d_in[0];
  const float* Wq = (const float*)d_in[1];
  const float* Wk = (const float*)d_in[2];
  const float* Wv = (const float*)d_in[3];
  float* out = (float*)d_out;

  // Workspace: t, Q, K, V in f16 (4 MB each, 16 MB total)
  size_t per = (size_t)B_ * N_ * C_;
  _Float16* t_h = (_Float16*)d_ws;
  _Float16* Qh  = t_h + per;
  _Float16* Kh  = Qh + per;
  _Float16* Vh  = Kh + per;

  transpose_to_f16<<<dim3(N_/32, C_/32, B_), dim3(32, 8), 0, stream>>>(x, t_h);
  qkv_proj<<<dim3(N_/128, B_), dim3(256), 0, stream>>>(t_h, Wq, Wk, Wv, Qh, Kh, Vh);
  flash_attn<<<dim3(N_/128, B_), dim3(256), 0, stream>>>(Qh, Kh, Vh, out);
}
